// GCN_56375740727523
// MI455X (gfx1250) — compile-verified, matrix operand order checked by reading, not run on hardware
//
#include <hip/hip_runtime.h>
#include <hip/hip_bf16.h>

// GCN, 2 layers: h1 = relu(Agg(x@W1) + b1); out = log_softmax(Agg(h1@W2) + b2)
// Agg(h)[n] = sum_{e: dst=n} h[src_e]*dinv[src]*dinv[dst] + h[n]*dinv[n]^2
// GEMMs use v_wmma_f32_16x16x32_bf16 (fp32 accumulate); aggregation is
// L2-resident gather + global_atomic_add_f32 (the bandwidth-bound part).

typedef __attribute__((ext_vector_type(16))) __bf16 v16bf;
typedef __attribute__((ext_vector_type(8)))  float  v8f;

// ---------------- utility: fill ----------------
__global__ void k_fill(float* __restrict__ p, long n, float v) {
    long i = (long)blockIdx.x * 256 + threadIdx.x;
    if (i < n) p[i] = v;
}

// ---------------- degree count (deg pre-initialized to 1.0 = self loop) ----
__global__ void k_deg(const int* __restrict__ ei, float* __restrict__ deg, int E) {
    int e = blockIdx.x * 256 + threadIdx.x;
    if (e < E) atomicAdd(&deg[ei[E + e]], 1.0f);
}

__global__ void k_rsqrt(const float* __restrict__ deg, float* __restrict__ dinv, int n) {
    int i = blockIdx.x * 256 + threadIdx.x;
    if (i < n) dinv[i] = rsqrtf(deg[i]);   // deg >= 1 always (self loop)
}

// ---------------- pack W1 (64x64) and W2 (64x40, padded to 48 cols) into the
// CDNA5 16-bit B-matrix lane/VGPR striping (ISA 7.12.2):
//   fragment f = ntile*2 + kchunk ; element i of lane L holds
//   K = kchunk*32 + (L>>4)*8 + (i<8 ? i : i+8), N = ntile*16 + (L&15)
__global__ void k_pack(const float* __restrict__ W1, const float* __restrict__ W2,
                       __bf16* __restrict__ w1p, __bf16* __restrict__ w2p) {
    int idx = blockIdx.x * 256 + threadIdx.x;
    if (idx < 8 * 512) {                       // W1: 4 ntiles * 2 kchunks
        int i = idx & 15, lane = (idx >> 4) & 31, f = idx >> 9;
        int kc = f & 1, nt = f >> 1;
        int K = kc * 32 + ((lane >> 4) << 3) + (i < 8 ? i : i + 8);
        int n = nt * 16 + (lane & 15);
        w1p[idx] = (__bf16)W1[K * 64 + n];
    } else if (idx < 8 * 512 + 6 * 512) {      // W2: 3 ntiles * 2 kchunks
        int j = idx - 8 * 512;
        int i = j & 15, lane = (j >> 4) & 31, f = j >> 9;
        int kc = f & 1, nt = f >> 1;
        int K = kc * 32 + ((lane >> 4) << 3) + (i < 8 ? i : i + 8);
        int n = nt * 16 + (lane & 15);
        w2p[j] = (n < 40) ? (__bf16)W2[K * 40 + n] : (__bf16)0.0f;
    }
}

// ---------------- WMMA GEMM: out[nrows x (NT*16)] = A[nrows x 64] * B -------
// One wave computes a 16-row strip across all NT column tiles; K=64 => 2 WMMA
// per tile. 8 waves per block => 128 rows per block. OS (output stride) is a
// template constant so row stores use immediate offsets off one base address.
template <int NT, int OS>
__global__ void k_gemm(const float* __restrict__ A, const __bf16* __restrict__ Bp,
                       float* __restrict__ out, int nrows) {
    const int lane = threadIdx.x & 31;
    const int wave = threadIdx.x >> 5;
    const int r0 = (blockIdx.x * 8 + wave) * 16;
    if (r0 >= nrows) return;

    // B fragments: packed so each lane reads 32 contiguous bytes per fragment
    v16bf bfr[NT * 2];
    const v16bf* bp = (const v16bf*)Bp;
#pragma unroll
    for (int f = 0; f < NT * 2; ++f) bfr[f] = bp[f * 32 + lane];

    // A fragments (16x32 bf16 each), built from fp32 rows on the fly.
    int m = r0 + (lane & 15);
    if (m > nrows - 1) m = nrows - 1;          // clamp; stores are guarded
    const float* row = A + (size_t)m * 64;
    const int kb = (lane >> 4) * 8;
    v16bf a0, a1;
#pragma unroll
    for (int j = 0; j < 8; ++j) {
        a0[j]     = (__bf16)row[kb + j];        // K = kb..kb+7
        a0[j + 8] = (__bf16)row[kb + 16 + j];   // K = kb+16..kb+23
        a1[j]     = (__bf16)row[kb + 32 + j];   // K = kb+32..kb+39
        a1[j + 8] = (__bf16)row[kb + 48 + j];   // K = kb+48..kb+55
    }

    const int rbase = r0 + ((lane >> 4) << 3);
    const bool full = (r0 + 15) < nrows;        // whole strip in range (common)
#pragma unroll
    for (int t = 0; t < NT; ++t) {
        v8f c = {};
        c = __builtin_amdgcn_wmma_f32_16x16x32_bf16(false, a0, false, bfr[t * 2 + 0],
                                                    (short)0, c, false, false);
        c = __builtin_amdgcn_wmma_f32_16x16x32_bf16(false, a1, false, bfr[t * 2 + 1],
                                                    (short)0, c, false, false);
        const int col = t * 16 + (lane & 15);
        float* ob = out + (size_t)rbase * OS + col;
        if (full) {
#pragma unroll
            for (int r = 0; r < 8; ++r) ob[r * OS] = c[r];
        } else {
#pragma unroll
            for (int r = 0; r < 8; ++r)
                if (rbase + r < nrows) ob[r * OS] = c[r];
        }
    }
}

// ---------------- edge aggregation, 64 feats: two edges per wave -----------
// 16 lanes x float4 = one 64-float row per half-wave (b128 gathers).
__global__ void k_agg64(const int* __restrict__ ei, const float* __restrict__ dinv,
                        const float* __restrict__ h, float* __restrict__ agg, int E) {
    int wv = blockIdx.x * 8 + (threadIdx.x >> 5);
    int e = wv * 2 + ((threadIdx.x >> 4) & 1);
    if (e >= E) return;
    int l = threadIdx.x & 15;
    int s = ei[e], d = ei[E + e];
    float nm = dinv[s] * dinv[d];
    float4 v = ((const float4*)(h + (size_t)s * 64))[l];
    float* ap = agg + (size_t)d * 64 + l * 4;
    atomicAdd(ap + 0, v.x * nm);
    atomicAdd(ap + 1, v.y * nm);
    atomicAdd(ap + 2, v.z * nm);
    atomicAdd(ap + 3, v.w * nm);
}

// ---------------- edge aggregation, 40 feats (input stride 48) -------------
__global__ void k_agg40(const int* __restrict__ ei, const float* __restrict__ dinv,
                        const float* __restrict__ g, float* __restrict__ agg, int E) {
    int w = blockIdx.x * 8 + (threadIdx.x >> 5);
    if (w >= E) return;
    int lane = threadIdx.x & 31;
    if (lane >= 20) return;
    int s = ei[w], d = ei[E + w];
    float nm = dinv[s] * dinv[d];
    float2 v = ((const float2*)(g + (size_t)s * 48))[lane];
    float* ap = agg + (size_t)d * 40 + lane * 2;
    atomicAdd(ap,     v.x * nm);
    atomicAdd(ap + 1, v.y * nm);
}

// ---------------- finalize layer1: + self-loop + bias, ReLU (in place) -----
__global__ void k_fin1(float* __restrict__ agg, const float* __restrict__ h,
                       const float* __restrict__ dinv, const float* __restrict__ b1,
                       long total) {
    long i = (long)blockIdx.x * 256 + threadIdx.x;
    if (i >= total) return;
    long n = i >> 6; int f = (int)(i & 63);
    float self = dinv[n] * dinv[n];
    float v = agg[i] + h[i] * self + b1[f];
    agg[i] = v > 0.0f ? v : 0.0f;
}

// ---------------- finalize layer2 + log_softmax: one wave per node ---------
__global__ void k_fin2(const float* __restrict__ agg2, const float* __restrict__ g,
                       const float* __restrict__ dinv, const float* __restrict__ b2,
                       float* __restrict__ out, int N) {
    int n = blockIdx.x * 8 + (threadIdx.x >> 5);
    if (n >= N) return;
    int lane = threadIdx.x & 31;
    float self = dinv[n] * dinv[n];
    float v0 = agg2[(size_t)n * 40 + lane] + g[(size_t)n * 48 + lane] * self + b2[lane];
    float v1 = -1e30f;
    if (lane < 8)
        v1 = agg2[(size_t)n * 40 + 32 + lane] + g[(size_t)n * 48 + 32 + lane] * self + b2[32 + lane];
    float mx = fmaxf(v0, v1);
#pragma unroll
    for (int o = 16; o; o >>= 1) mx = fmaxf(mx, __shfl_xor(mx, o, 32));
    float s = __expf(v0 - mx) + (lane < 8 ? __expf(v1 - mx) : 0.0f);
#pragma unroll
    for (int o = 16; o; o >>= 1) s += __shfl_xor(s, o, 32);
    float ls = __logf(s);
    out[(size_t)n * 40 + lane] = v0 - mx - ls;
    if (lane < 8) out[(size_t)n * 40 + 32 + lane] = v1 - mx - ls;
}

// ============================ host launcher ================================
extern "C" void kernel_launch(void* const* d_in, const int* in_sizes, int n_in,
                              void* d_out, int out_size, void* d_ws, size_t ws_size,
                              hipStream_t stream) {
    const float* x  = (const float*)d_in[0];
    const int*   ei = (const int*)d_in[1];   // harness maps integer inputs -> int32
    const float* W1 = (const float*)d_in[2];
    const float* b1 = (const float*)d_in[3];
    const float* W2 = (const float*)d_in[4];
    const float* b2 = (const float*)d_in[5];
    float* out = (float*)d_out;

    const long N = in_sizes[0] / 64;
    const int  E = in_sizes[1] / 2;

    char* ws = (char*)d_ws;
    size_t off = 0;
    auto carve = [&](size_t bytes) -> char* {
        char* p = ws + off;
        off += (bytes + 255) & ~(size_t)255;
        return p;
    };
    float*  deg  = (float*)carve(N * 4);
    float*  dinv = (float*)carve(N * 4);
    float*  h1   = (float*)carve(N * 64 * 4);   // layer1 transform; reused as g (stride 48)
    float*  agg1 = (float*)carve(N * 64 * 4);   // layer1 accum -> relu'd hidden
    float*  agg2 = (float*)carve(N * 40 * 4);   // layer2 accum
    __bf16* w1p  = (__bf16*)carve(8 * 512 * 2);
    __bf16* w2p  = (__bf16*)carve(6 * 512 * 2);
    float*  g    = h1;                          // reuse after layer1 finalize

    const int T = 256;
    // zero/init accumulators every call (graph-replay safe)
    k_fill<<<(int)((N + T - 1) / T), T, 0, stream>>>(deg, N, 1.0f);          // self loop
    k_fill<<<(int)((N * 64 + T - 1) / T), T, 0, stream>>>(agg1, N * 64, 0.0f);
    k_fill<<<(int)((N * 40 + T - 1) / T), T, 0, stream>>>(agg2, N * 40, 0.0f);

    k_deg<<<(E + T - 1) / T, T, 0, stream>>>(ei, deg, E);
    k_rsqrt<<<(int)((N + T - 1) / T), T, 0, stream>>>(deg, dinv, (int)N);
    k_pack<<<(8 * 512 + 6 * 512 + T - 1) / T, T, 0, stream>>>(W1, W2, w1p, w2p);

    const int gemmBlocks = (int)((N + 127) / 128);

    // Layer 1
    k_gemm<4, 64><<<gemmBlocks, T, 0, stream>>>(x, w1p, h1, (int)N);
    k_agg64<<<(E + 15) / 16, T, 0, stream>>>(ei, dinv, h1, agg1, E);
    k_fin1<<<(int)((N * 64 + T - 1) / T), T, 0, stream>>>(agg1, h1, dinv, b1, N * 64);

    // Layer 2 (output cols padded 40->48 in g for aligned float2 access)
    k_gemm<3, 48><<<gemmBlocks, T, 0, stream>>>(agg1, w2p, g, (int)N);
    k_agg40<<<(E + 7) / 8, T, 0, stream>>>(ei, dinv, g, agg2, E);
    k_fin2<<<(int)((N + 7) / 8), T, 0, stream>>>(agg2, g, dinv, b2, out, (int)N);
}
